// TransformerLLM_76003741270438
// MI455X (gfx1250) — compile-verified, matrix operand order checked by reading, not run on hardware
//
#include <hip/hip_runtime.h>

// ---------------------------------------------------------------------------
// Problem constants (match reference)
// ---------------------------------------------------------------------------
#define BB 2
#define SS 1024
#define DD 1024
#define HH 16
#define LL 4
#define FF 4096
#define VV 32000
#define TT (BB * SS)     // 2048 tokens
#define DK 64            // head dim

typedef __attribute__((ext_vector_type(16))) _Float16 v16h;
typedef __attribute__((ext_vector_type(8)))  float    v8f;
typedef __attribute__((ext_vector_type(4)))  unsigned v4u;
typedef __attribute__((ext_vector_type(4)))  int      v4i;
typedef __attribute__((ext_vector_type(8)))  int      v8i;

#if __has_builtin(__builtin_amdgcn_tensor_load_to_lds)
#define USE_TDM 1
#else
#define USE_TDM 0
#endif

// ---------------------------------------------------------------------------
// WMMA helpers (CDNA5: V_WMMA_F32_16X16X32_F16, wave32)
// A/B operand layout (16-bit, 16x32): lane L -> row/col = L%16, g = L/16,
// halves 0..7 hold K = g*8+0..7, halves 8..15 hold K = 16+g*8+0..7.
// ---------------------------------------------------------------------------
__device__ __forceinline__ v8f wmma_f16(v16h a, v16h b, v8f c) {
  return __builtin_amdgcn_wmma_f32_16x16x32_f16(false, a, false, b, (short)0, c,
                                                false, false);
}

__device__ __forceinline__ v16h frag_ld(const _Float16* rowp, int g) {
  v16h r;
#pragma unroll
  for (int j = 0; j < 8; ++j) {
    r[j]     = rowp[g * 8 + j];
    r[8 + j] = rowp[16 + g * 8 + j];
  }
  return r;
}

// ---------------------------------------------------------------------------
// Embedding gather + sinusoidal positional encoding -> f32 residual stream
// ---------------------------------------------------------------------------
__global__ __launch_bounds__(256) void embed_kernel(
    const int* __restrict__ ids, const float* __restrict__ emb,
    float* __restrict__ x) {
  const int t   = blockIdx.x;
  const int pos = t % SS;
  const int id  = ids[t];
  const int d0  = threadIdx.x * 4;
  const float4 e = ((const float4*)(emb + (size_t)id * DD))[threadIdx.x];
  float ev[4] = {e.x, e.y, e.z, e.w};
  const float kln = 9.2103403719761836f / (float)DD;  // ln(10000)/D
#pragma unroll
  for (int j = 0; j < 4; ++j) {
    int d = d0 + j;
    float freq = __expf(-(float)(d & ~1) * kln);
    float ang  = (float)pos * freq;
    float pe   = (d & 1) ? __cosf(ang) : __sinf(ang);
    x[(size_t)t * DD + d] = ev[j] + pe;
  }
}

// ---------------------------------------------------------------------------
// LayerNorm: one block per token row, D=1024, emits f16 (GEMM A operand)
// ---------------------------------------------------------------------------
__global__ __launch_bounds__(256) void ln_kernel(
    const float* __restrict__ x, const float* __restrict__ w,
    const float* __restrict__ b, _Float16* __restrict__ out) {
  const int row = blockIdx.x;
  const int tid = threadIdx.x;
  const float4 v = ((const float4*)(x + (size_t)row * DD))[tid];
  __shared__ float rs[256], rq[256];
  rs[tid] = v.x + v.y + v.z + v.w;
  rq[tid] = v.x * v.x + v.y * v.y + v.z * v.z + v.w * v.w;
  __syncthreads();
#pragma unroll
  for (int off = 128; off > 0; off >>= 1) {
    if (tid < off) { rs[tid] += rs[tid + off]; rq[tid] += rq[tid + off]; }
    __syncthreads();
  }
  const float mean = rs[0] * (1.0f / DD);
  const float var  = rq[0] * (1.0f / DD) - mean * mean;
  const float rstd = rsqrtf(var + 1e-5f);
  const int d0 = tid * 4;
  float vv[4] = {v.x, v.y, v.z, v.w};
#pragma unroll
  for (int j = 0; j < 4; ++j) {
    int d = d0 + j;
    out[(size_t)row * DD + d] =
        (_Float16)((vv[j] - mean) * rstd * w[d] + b[d]);
  }
}

// ---------------------------------------------------------------------------
// Tiled WMMA GEMM: C[M,N] = A[M,K](f16) * B[K,N](f32, converted in-flight)
// Block = 256 threads = 8 waves; block tile 128x128, K-step 32.
// Wave grid 4x2: each wave computes a 32x64 slab = 2x4 WMMA accumulators.
// A tile (128x32 f16, row-major, stride K) is DMA'd by the Tensor Data Mover
// (one tensor_load_to_lds per K-step, issued by wave 0) while all 8 waves
// stage + convert the f32 B tile; then s_wait_tensorcnt 0 + barrier.
// MODE: 0 = store f16, 1 = +bias, exact GELU, store f16,
//       2 = (+bias) + residual, store f32, 3 = plain f32 store.
// BT:   B is stored transposed (N x K row-major), i.e. B[k][n] = Bg[n*ldb+k].
// ---------------------------------------------------------------------------
template <int MODE, bool BT>
__global__ __launch_bounds__(256) void gemm_kernel(
    const _Float16* __restrict__ A, const float* __restrict__ Bg,
    const float* __restrict__ bias, const float* __restrict__ resid,
    float* __restrict__ outF, _Float16* __restrict__ outH,
    int M, int N, int K, int ldb) {
  __shared__ _Float16 As[128][32];  // [m][k]
  __shared__ _Float16 Bt[128][32];  // [n][k] (transposed for frag loads)
  const int tid  = threadIdx.x;
  const int lane = tid & 31, wave = tid >> 5;
  const int m0 = blockIdx.y * 128, n0 = blockIdx.x * 128;
  const int mo = (wave >> 1) * 32, no = (wave & 1) * 64;
  const int lm = lane & 15, g = lane >> 4;
  v8f c[2][4] = {};

#if USE_TDM
  // ---- Tensor DMA descriptor, group 1 (loop-invariant): 2-byte elements,
  // tensor K x M with row stride K, tile 32(K) x 128(rows) -----------------
  const unsigned ldsA = (unsigned)(uintptr_t)&As[0][0];  // LDS byte address
  v8i g1;
  g1[0] = (int)(1u << 16);                                   // data_size = 2B
  g1[1] = (int)(((unsigned)K & 0xFFFFu) << 16);              // tensor_dim0 lo
  g1[2] = (int)(((unsigned)K >> 16) | (((unsigned)M & 0xFFFFu) << 16));
  g1[3] = (int)(((unsigned)M >> 16) | (32u << 16));          // tile_dim0 = 32
  g1[4] = 128;                                               // tile_dim1 = 128
  g1[5] = K;                                                 // dim0_stride
  g1[6] = 0;
  g1[7] = 0;
  const v4i z4 = {};
#if __clang_major__ >= 23
  const v8i z8 = {};
#endif
#endif

  for (int k0 = 0; k0 < K; k0 += 32) {
#if USE_TDM
    if (wave == 0) {  // one TDM op per block covers the whole 128x32 A tile
      const uintptr_t ga = (uintptr_t)(A + (size_t)m0 * K + k0);
      v4u g0;
      g0[0] = 1u;                                            // count = 1
      g0[1] = ldsA;                                          // lds_addr
      g0[2] = (unsigned)ga;                                  // addr[31:0]
      g0[3] = (unsigned)((ga >> 32) & 0x01FFFFFFu) | (2u << 30);  // type=2
#if __clang_major__ >= 23
      __builtin_amdgcn_tensor_load_to_lds(g0, g1, z4, z4, z8, 0);
#else
      __builtin_amdgcn_tensor_load_to_lds(g0, g1, z4, z4, 0);
#endif
    }
#else
    {  // manual A staging: 16 halves (32B) per thread
      const int row = tid >> 1, kg = tid & 1;
      const uint4* sp = (const uint4*)(A + (size_t)(m0 + row) * K + k0 + kg * 16);
      uint4 a0 = sp[0], a1 = sp[1];
      *(uint4*)&As[row][kg * 16 + 0] = a0;
      *(uint4*)&As[row][kg * 16 + 8] = a1;
    }
#endif
    // ---- stage B (f32 -> f16), 16 floats per thread ----------------------
    if (BT) {  // B rows are N-major: contiguous along K -> contiguous store
      const int n = tid >> 1, kg = tid & 1;
      const float4* sp =
          (const float4*)(Bg + (size_t)(n0 + n) * ldb + k0 + kg * 16);
      float4 f[4] = {sp[0], sp[1], sp[2], sp[3]};
      const float* fp = (const float*)f;
#pragma unroll
      for (int j = 0; j < 16; ++j) Bt[n][kg * 16 + j] = (_Float16)fp[j];
      if (k0 + 32 < K)
        __builtin_prefetch(Bg + (size_t)(n0 + n) * ldb + k0 + 32 + kg * 16, 0, 1);
    } else {   // B is KxN: contiguous along N -> transposing scatter to LDS
      const int kk = tid >> 3, ng = tid & 7;
      const float4* sp =
          (const float4*)(Bg + (size_t)(k0 + kk) * ldb + n0 + ng * 16);
      float4 f[4] = {sp[0], sp[1], sp[2], sp[3]};
      const float* fp = (const float*)f;
#pragma unroll
      for (int j = 0; j < 16; ++j) Bt[ng * 16 + j][kk] = (_Float16)fp[j];
      if (k0 + 32 < K)
        __builtin_prefetch(Bg + (size_t)(k0 + 32 + kk) * ldb + n0 + ng * 16, 0, 1);
    }
#if USE_TDM
    if (wave == 0) __builtin_amdgcn_s_wait_tensorcnt(0);
#endif
    __syncthreads();
    // ---- 2x4 WMMA tiles per wave ----------------------------------------
    const v16h a0 = frag_ld(&As[mo + lm][0], g);
    const v16h a1 = frag_ld(&As[mo + 16 + lm][0], g);
#pragma unroll
    for (int t = 0; t < 4; ++t) {
      const v16h b = frag_ld(&Bt[no + t * 16 + lm][0], g);
      c[0][t] = wmma_f16(a0, b, c[0][t]);
      c[1][t] = wmma_f16(a1, b, c[1][t]);
    }
    __syncthreads();
  }

  // ---- Epilogue. C layout: (m,n) -> lane = n%16 + 16*(m/8), vgpr = m%8 ---
#pragma unroll
  for (int mi = 0; mi < 2; ++mi) {
#pragma unroll
    for (int r = 0; r < 8; ++r) {
      const int m = m0 + mo + mi * 16 + r + 8 * g;
#pragma unroll
      for (int t = 0; t < 4; ++t) {
        const int n = n0 + no + t * 16 + lm;
        float v = c[mi][t][r];
        const size_t idx = (size_t)m * N + n;
        if (MODE == 1) {
          v += bias[n];
          v = 0.5f * v * (1.0f + erff(v * 0.70710678118654752f));  // exact GELU
          outH[idx] = (_Float16)v;
        } else if (MODE == 0) {
          outH[idx] = (_Float16)v;
        } else if (MODE == 2) {
          if (bias) v += bias[n];
          outF[idx] = resid[idx] + v;
        } else {
          outF[idx] = v;
        }
      }
    }
  }
}

// ---------------------------------------------------------------------------
// Flash attention: one wave per (b, h, 16-query tile). Causal, scale 1/8.
// Scores via WMMA straight from global f16 Q/K (rows are contiguous 64 halves
// = exact A/B operand layout). Online softmax with half-wave shfl reductions
// (wave32: xor masks 1/2/4/8 stay inside the 16-lane C-row groups).
// P repacked through LDS into A-operand layout; P*V via 4 WMMAs, V staged
// transposed ([dim][key]) in LDS.
// ---------------------------------------------------------------------------
__global__ __launch_bounds__(32) void attn_kernel(
    const _Float16* __restrict__ q, const _Float16* __restrict__ k,
    const _Float16* __restrict__ v, _Float16* __restrict__ o) {
  const int lane = threadIdx.x;
  const int lm = lane & 15, g = lane >> 4;
  const int qbase = blockIdx.x * 16;
  const int h = blockIdx.y, b = blockIdx.z;
  const size_t tb = (size_t)b * SS;
  const int hd = h * DK;

  __shared__ _Float16 Pld[16][32];  // probs, [m][key] (A-operand layout)
  __shared__ _Float16 Vt[64][32];   // V tile transposed, [dim][key]

  const _Float16* qrow = q + (tb + qbase + lm) * DD + hd;
  const v16h qa0 = frag_ld(qrow, g);       // K-dims 0..31
  const v16h qa1 = frag_ld(qrow + 32, g);  // K-dims 32..63

  float mi[8], li[8];
  v8f oc0 = {}, oc1 = {}, oc2 = {}, oc3 = {};
#pragma unroll
  for (int r = 0; r < 8; ++r) { mi[r] = -1e30f; li[r] = 0.0f; }

  const int nch = (qbase + 15) / 32 + 1;  // causal: chunks of 32 keys
  for (int ch = 0; ch < nch; ++ch) {
    const int kv0 = ch * 32;
    v8f s0 = {}, s1 = {};
    {
      const _Float16* kr0 = k + (tb + kv0 + lm) * DD + hd;
      s0 = wmma_f16(qa0, frag_ld(kr0, g), s0);
      s0 = wmma_f16(qa1, frag_ld(kr0 + 32, g), s0);
      const _Float16* kr1 = k + (tb + kv0 + 16 + lm) * DD + hd;
      s1 = wmma_f16(qa0, frag_ld(kr1, g), s1);
      s1 = wmma_f16(qa1, frag_ld(kr1 + 32, g), s1);
    }
#pragma unroll
    for (int r = 0; r < 8; ++r) {
      const int m = qbase + r + 8 * g;
      const int k0i = kv0 + lm, k1i = kv0 + 16 + lm;
      float a0 = (k0i <= m) ? s0[r] * 0.125f : -1e30f;
      float a1 = (k1i <= m) ? s1[r] * 0.125f : -1e30f;
      float rm = fmaxf(a0, a1);
      rm = fmaxf(rm, __shfl_xor(rm, 1, 32));
      rm = fmaxf(rm, __shfl_xor(rm, 2, 32));
      rm = fmaxf(rm, __shfl_xor(rm, 4, 32));
      rm = fmaxf(rm, __shfl_xor(rm, 8, 32));
      const float mn = fmaxf(mi[r], rm);
      const float al = __expf(mi[r] - mn);
      a0 = __expf(a0 - mn);
      a1 = __expf(a1 - mn);
      float rs = a0 + a1;
      rs += __shfl_xor(rs, 1, 32);
      rs += __shfl_xor(rs, 2, 32);
      rs += __shfl_xor(rs, 4, 32);
      rs += __shfl_xor(rs, 8, 32);
      li[r] = li[r] * al + rs;
      mi[r] = mn;
      oc0[r] *= al; oc1[r] *= al; oc2[r] *= al; oc3[r] *= al;
      Pld[r + 8 * g][lm]      = (_Float16)a0;
      Pld[r + 8 * g][16 + lm] = (_Float16)a1;
    }
    {
      const _Float16* vr = v + (tb + kv0 + lane) * DD + hd;
#pragma unroll
      for (int d = 0; d < 64; ++d) Vt[d][lane] = vr[d];
    }
    __syncthreads();
    const v16h pa = frag_ld(&Pld[lm][0], g);
    oc0 = wmma_f16(pa, frag_ld(&Vt[lm][0], g), oc0);
    oc1 = wmma_f16(pa, frag_ld(&Vt[16 + lm][0], g), oc1);
    oc2 = wmma_f16(pa, frag_ld(&Vt[32 + lm][0], g), oc2);
    oc3 = wmma_f16(pa, frag_ld(&Vt[48 + lm][0], g), oc3);
    __syncthreads();
  }
#pragma unroll
  for (int r = 0; r < 8; ++r) {
    const float inv = 1.0f / li[r];
    const size_t trow = (tb + qbase + r + 8 * g) * DD + hd;
    o[trow + lm]      = (_Float16)(oc0[r] * inv);
    o[trow + 16 + lm] = (_Float16)(oc1[r] * inv);
    o[trow + 32 + lm] = (_Float16)(oc2[r] * inv);
    o[trow + 48 + lm] = (_Float16)(oc3[r] * inv);
  }
}

// ---------------------------------------------------------------------------
// Host orchestration
// ---------------------------------------------------------------------------
extern "C" void kernel_launch(void* const* d_in, const int* in_sizes, int n_in,
                              void* d_out, int out_size, void* d_ws,
                              size_t ws_size, hipStream_t stream) {
  (void)in_sizes; (void)n_in; (void)out_size; (void)ws_size;
  const int*   ids  = (const int*)d_in[0];
  const float* emb  = (const float*)d_in[1];
  const float* Wq   = (const float*)d_in[2];
  const float* Wk   = (const float*)d_in[3];
  const float* Wv   = (const float*)d_in[4];
  const float* Wo   = (const float*)d_in[5];
  const float* W1   = (const float*)d_in[6];
  const float* b1   = (const float*)d_in[7];
  const float* W2   = (const float*)d_in[8];
  const float* b2   = (const float*)d_in[9];
  const float* ln1w = (const float*)d_in[10];
  const float* ln1b = (const float*)d_in[11];
  const float* ln2w = (const float*)d_in[12];
  const float* ln2b = (const float*)d_in[13];
  const float* lnfw = (const float*)d_in[14];
  const float* lnfb = (const float*)d_in[15];
  float* logits = (float*)d_out;

  char* p = (char*)d_ws;
  auto carve = [&](size_t bytes) -> char* {
    char* r = p;
    p += (bytes + 255) & ~(size_t)255;
    return r;
  };
  float*    x  = (float*)carve((size_t)TT * DD * 4);     // residual stream f32
  _Float16* hh = (_Float16*)carve((size_t)TT * DD * 2);  // LN output f16
  _Float16* qb = (_Float16*)carve((size_t)TT * DD * 2);
  _Float16* kb = (_Float16*)carve((size_t)TT * DD * 2);
  _Float16* vb = (_Float16*)carve((size_t)TT * DD * 2);
  _Float16* ob = (_Float16*)carve((size_t)TT * DD * 2);  // attention output
  _Float16* ub = (_Float16*)carve((size_t)TT * FF * 2);  // FFN hidden f16

  embed_kernel<<<TT, 256, 0, stream>>>(ids, emb, x);

  const dim3 gD(DD / 128, TT / 128);   // (8, 16)
  const dim3 gF(FF / 128, TT / 128);   // (32, 16)
  const dim3 gV(VV / 128, TT / 128);   // (250, 16)
  const dim3 gA(SS / 16, HH, BB);      // (64, 16, 2)

  for (int l = 0; l < LL; ++l) {
    const size_t wD = (size_t)l * DD * DD;
    // --- attention block (pre-LN) ---
    ln_kernel<<<TT, 256, 0, stream>>>(x, ln1w + l * DD, ln1b + l * DD, hh);
    gemm_kernel<0, false><<<gD, 256, 0, stream>>>(hh, Wq + wD, nullptr, nullptr,
                                                  nullptr, qb, TT, DD, DD, DD);
    gemm_kernel<0, false><<<gD, 256, 0, stream>>>(hh, Wk + wD, nullptr, nullptr,
                                                  nullptr, kb, TT, DD, DD, DD);
    gemm_kernel<0, false><<<gD, 256, 0, stream>>>(hh, Wv + wD, nullptr, nullptr,
                                                  nullptr, vb, TT, DD, DD, DD);
    attn_kernel<<<gA, 32, 0, stream>>>(qb, kb, vb, ob);
    gemm_kernel<2, false><<<gD, 256, 0, stream>>>(ob, Wo + wD, nullptr, x, x,
                                                  nullptr, TT, DD, DD, DD);
    // --- FFN block (pre-LN) ---
    ln_kernel<<<TT, 256, 0, stream>>>(x, ln2w + l * DD, ln2b + l * DD, hh);
    gemm_kernel<1, false><<<gF, 256, 0, stream>>>(
        hh, W1 + (size_t)l * DD * FF, b1 + (size_t)l * FF, nullptr, nullptr,
        ub, TT, FF, DD, FF);
    gemm_kernel<2, false><<<gD, 256, 0, stream>>>(
        ub, W2 + (size_t)l * FF * DD, b2 + (size_t)l * DD, x, x, nullptr, TT,
        DD, FF, DD);
  }

  // final LN + tied-embedding logits (B accessed transposed: emb is V x D)
  ln_kernel<<<TT, 256, 0, stream>>>(x, lnfw, lnfb, hh);
  gemm_kernel<3, true><<<gV, 256, 0, stream>>>(hh, emb, nullptr, nullptr,
                                               logits, nullptr, TT, VV, DD, DD);
}
